// disciminativeAno_58866821759213
// MI455X (gfx1250) — matrix operand order repaired and checked by reading, not compiled, not run on hardware
//
#include <hip/hip_runtime.h>
#include <hip/hip_bf16.h>
#include <math.h>

typedef __attribute__((ext_vector_type(16))) _Float16 v16h;
typedef __attribute__((ext_vector_type(8)))  _Float16 v8h;
typedef __attribute__((ext_vector_type(8)))  float    v8f;

#define NBINS 101

// ===========================================================================
// Weight transpose+pad: w f32 [K,N] (flattened HWIO) -> bt f16 [Npad, Kpad]
// ===========================================================================
__global__ void wtrans_kernel(const float* __restrict__ w,
                              _Float16* __restrict__ bt,
                              int K, int N, int Kpad, int Npad) {
  int idx = blockIdx.x * blockDim.x + threadIdx.x;
  if (idx >= Npad * Kpad) return;
  int k = idx % Kpad;
  int n = idx / Kpad;
  _Float16 v = (_Float16)0.0f;
  if (k < K && n < N) v = (_Float16)w[(size_t)k * N + n];
  bt[idx] = v;
}

// ===========================================================================
// Conv1 im2col (Cin=3, k=5, s=2, pad=2, 32x32 -> 16x16), fused f32->f16.
// Output A1 [M, 96] zero-padded in K (75 -> 96).
// ===========================================================================
__global__ void im2col_conv1_kernel(const float* __restrict__ x,
                                    _Float16* __restrict__ a, int total) {
  int idx = blockIdx.x * blockDim.x + threadIdx.x;
  if (idx >= total) return;
  int k = idx % 96;
  int m = idx / 96;
  _Float16 v = (_Float16)0.0f;
  if (k < 75) {
    int c = k % 3, t = k / 3;
    int kx = t % 5, ky = t / 5;
    int ox = m & 15, oy = (m >> 4) & 15, b = m >> 8;
    int iy = oy * 2 + ky - 2, ix = ox * 2 + kx - 2;
    if (iy >= 0 && iy < 32 && ix >= 0 && ix < 32)
      v = (_Float16)x[(((size_t)b * 32 + iy) * 32 + ix) * 3 + c];
  }
  a[idx] = v;
}

// ===========================================================================
// Fragment helpers (ISA 7.12.2 layouts, wave32).
// ===========================================================================
__device__ __forceinline__ v16h cat8(v8h lo, v8h hi) {
  return __builtin_shufflevector(lo, hi, 0, 1, 2, 3, 4, 5, 6, 7,
                                 8, 9, 10, 11, 12, 13, 14, 15);
}

__device__ __forceinline__ float act_bn(float v, const int ACT, const int BN,
                                        float g, float be) {
  if (ACT == 1) v = fmaxf(v, 0.0f);
  else if (ACT == 2) v = 1.0f / (1.0f + __expf(-v));
  if (BN) v = v * g + be;
  return v;
}

// ===========================================================================
// Dense WMMA GEMM, ping-pong double-buffered.
//   A [M,K] f16 row-major, Bt [Npad,K] f16 row-major, K % 32 == 0.
//   Each wave computes (16*MT) x (16*NT) of C.
// ===========================================================================
template <int MT, int NT, int ACT, int HAS_BN>
__global__ __launch_bounds__(256)
void wmma_gemm_tpl(const _Float16* __restrict__ A,
                   const _Float16* __restrict__ Bt,
                   const float* __restrict__ bias,
                   const float* __restrict__ gamma,
                   const float* __restrict__ beta,
                   _Float16* __restrict__ out,
                   int M, int K, int Nreal) {
  const int wave = threadIdx.x >> 5;
  const int lane = threadIdx.x & 31;
  const int tile0 = (blockIdx.x * 8 + wave) * MT;
  if (tile0 * 16 >= M) return;

  const int lhi = lane >> 4;
  const int a_kb = lhi * 8;
  const int b_kb = lhi * 8;
  const int ncol0 = blockIdx.y * (16 * NT) + (lane & 15);

  const _Float16* arow[MT];
#pragma unroll
  for (int mt = 0; mt < MT; ++mt)
    arow[mt] = A + (size_t)((tile0 + mt) * 16 + (lane & 15)) * K;
  const _Float16* btrow[NT];
#pragma unroll
  for (int t = 0; t < NT; ++t)
    btrow[t] = Bt + (size_t)(ncol0 + t * 16) * K;

  v8f acc[MT][NT];
#pragma unroll
  for (int mt = 0; mt < MT; ++mt)
#pragma unroll
    for (int t = 0; t < NT; ++t) acc[mt][t] = (v8f){};

  auto loadA = [&](int k0, v16h* a) {
#pragma unroll
    for (int mt = 0; mt < MT; ++mt)
      a[mt] = cat8(*(const v8h*)(arow[mt] + k0 + a_kb),
                   *(const v8h*)(arow[mt] + k0 + a_kb + 16));
  };
  auto loadB = [&](int k0, v16h* b) {
#pragma unroll
    for (int t = 0; t < NT; ++t)
      b[t] = cat8(*(const v8h*)(btrow[t] + k0 + b_kb),
                  *(const v8h*)(btrow[t] + k0 + b_kb + 16));
  };
  auto step = [&](v16h* a, v16h* b) {
#pragma unroll
    for (int mt = 0; mt < MT; ++mt)
#pragma unroll
      for (int t = 0; t < NT; ++t)
        acc[mt][t] = __builtin_amdgcn_wmma_f32_16x16x32_f16(
            false, a[mt], false, b[t], (short)0, acc[mt][t], false, false);
  };

  v16h aA[MT], bA[NT], aB[MT], bB[NT];
  loadA(0, aA);
  loadB(0, bA);
  for (int k0 = 0; k0 < K; k0 += 64) {
    // speculative prefetch: safe even past end (dropped on xlat fail)
    __builtin_prefetch(arow[0] + k0 + 512, 0, 1);
    if (k0 + 32 < K) { loadA(k0 + 32, aB); loadB(k0 + 32, bB); }
    step(aA, bA);
    if (k0 + 32 < K) {
      if (k0 + 64 < K) { loadA(k0 + 64, aA); loadB(k0 + 64, bA); }
      step(aB, bB);
    }
  }

#pragma unroll
  for (int t = 0; t < NT; ++t) {
    int n = ncol0 + t * 16;
    if (n < Nreal) {
      const float bs = bias[n];
      const float g  = HAS_BN ? gamma[n] : 1.0f;
      const float be = HAS_BN ? beta[n]  : 0.0f;
#pragma unroll
      for (int mt = 0; mt < MT; ++mt)
#pragma unroll
        for (int r = 0; r < 8; ++r) {
          int m = (tile0 + mt) * 16 + r + lhi * 8;
          float v = act_bn(acc[mt][t][r] + bs, ACT, HAS_BN, g, be);
          out[(size_t)m * Nreal + n] = (_Float16)v;
        }
    }
  }
}

// ===========================================================================
// Implicit-GEMM conv / conv-transpose, static geometry, flattened K-chunk
// loop with ping-pong double buffering. CIN % 32 == 0.
//   MODE 0: conv      iy = oy*STRIDE + ky - PAD
//   MODE 1: convT     iy = (oy + PAD - ky)/STRIDE when divisible
// ===========================================================================
template <int HIN, int WIN, int CIN, int KH, int KW, int HO, int WO,
          int STRIDE, int PAD, int MODE, int MT, int NT, int ACT, int HAS_BN>
__global__ __launch_bounds__(256)
void wmma_conv_tpl(const _Float16* __restrict__ in,
                   const _Float16* __restrict__ Bt,   // [Npad, K]
                   const float* __restrict__ bias,
                   const float* __restrict__ gamma,
                   const float* __restrict__ beta,
                   _Float16* __restrict__ out,
                   int Bn, int Nreal) {
  constexpr int K = KH * KW * CIN;
  constexpr int CPT = CIN / 32;       // K-chunks per (ky,kx) tap
  constexpr int NCHUNK = K / 32;
  const int M = Bn * HO * WO;

  const int wave = threadIdx.x >> 5;
  const int lane = threadIdx.x & 31;
  const int tile0 = (blockIdx.x * 8 + wave) * MT;
  if (tile0 * 16 >= M) return;

  const int lhi = lane >> 4;
  const int a_kb = lhi * 8;
  const int b_kb = lhi * 8;
  const int ncol0 = blockIdx.y * (16 * NT) + (lane & 15);

  int ox[MT], oy[MT];
  const _Float16* img[MT];
#pragma unroll
  for (int mt = 0; mt < MT; ++mt) {
    int mrow = (tile0 + mt) * 16 + (lane & 15);
    ox[mt] = mrow % WO;                 // power-of-two -> masks/shifts
    oy[mt] = (mrow / WO) % HO;
    img[mt] = in + (size_t)(mrow / (WO * HO)) * HIN * WIN * CIN;
  }

  const _Float16* btrow[NT];
#pragma unroll
  for (int t = 0; t < NT; ++t)
    btrow[t] = Bt + (size_t)(ncol0 + t * 16) * K;

  v8f acc[MT][NT];
#pragma unroll
  for (int mt = 0; mt < MT; ++mt)
#pragma unroll
    for (int t = 0; t < NT; ++t) acc[mt][t] = (v8f){};

  auto loadA = [&](int q, v16h* a) {
    const int tap = q / CPT;            // uniform -> SALU
    const int c0  = (q % CPT) * 32;
    const int ky = tap / KW, kx = tap % KW;
#pragma unroll
    for (int mt = 0; mt < MT; ++mt) {
      int iy, ix;
      bool ok;
      if (MODE == 0) {
        iy = oy[mt] * STRIDE + ky - PAD;
        ix = ox[mt] * STRIDE + kx - PAD;
        ok = (iy >= 0) && (iy < HIN) && (ix >= 0) && (ix < WIN);
      } else {
        int ty = oy[mt] + PAD - ky;
        int tx = ox[mt] + PAD - kx;
        ok = (ty >= 0) && (tx >= 0) &&
             ((ty % STRIDE) == 0) && ((tx % STRIDE) == 0);
        iy = ty / STRIDE;
        ix = tx / STRIDE;
        ok = ok && (iy < HIN) && (ix < WIN);
      }
      const _Float16* src = img[mt] + ((size_t)iy * WIN + ix) * CIN + c0;
      v8h a0 = {}, a1 = {};
      if (ok) {
        a0 = *(const v8h*)(src + a_kb);
        a1 = *(const v8h*)(src + a_kb + 16);
      }
      a[mt] = cat8(a0, a1);
    }
  };
  auto loadB = [&](int q, v16h* b) {
    const int k = q * 32 + b_kb;
#pragma unroll
    for (int t = 0; t < NT; ++t)
      b[t] = cat8(*(const v8h*)(btrow[t] + k),
                  *(const v8h*)(btrow[t] + k + 16));
  };
  auto step = [&](v16h* a, v16h* b) {
#pragma unroll
    for (int mt = 0; mt < MT; ++mt)
#pragma unroll
      for (int t = 0; t < NT; ++t)
        acc[mt][t] = __builtin_amdgcn_wmma_f32_16x16x32_f16(
            false, a[mt], false, b[t], (short)0, acc[mt][t], false, false);
  };

  v16h aA[MT], bA[NT], aB[MT], bB[NT];
  loadA(0, aA);
  loadB(0, bA);
  for (int q = 0; q < NCHUNK; q += 2) {
    if (q + 1 < NCHUNK) { loadA(q + 1, aB); loadB(q + 1, bB); }
    step(aA, bA);
    if (q + 1 < NCHUNK) {
      if (q + 2 < NCHUNK) { loadA(q + 2, aA); loadB(q + 2, bA); }
      step(aB, bB);
    }
  }

#pragma unroll
  for (int t = 0; t < NT; ++t) {
    int n = ncol0 + t * 16;
    if (n < Nreal) {
      const float bs = bias[n];
      const float g  = HAS_BN ? gamma[n] : 1.0f;
      const float be = HAS_BN ? beta[n]  : 0.0f;
#pragma unroll
      for (int mt = 0; mt < MT; ++mt)
#pragma unroll
        for (int r = 0; r < 8; ++r) {
          int m = (tile0 + mt) * 16 + r + lhi * 8;
          float v = act_bn(acc[mt][t][r] + bs, ACT, HAS_BN, g, be);
          out[(size_t)m * Nreal + n] = (_Float16)v;
        }
    }
  }
}

// ===========================================================================
// Sampling: z = z_mean + exp(0.5 * z_log_var) * eps
// ===========================================================================
__global__ void sampling_kernel(const _Float16* __restrict__ zm,
                                const _Float16* __restrict__ zlv,
                                const float* __restrict__ eps,
                                _Float16* __restrict__ z, int n) {
  int i = blockIdx.x * blockDim.x + threadIdx.x;
  if (i < n) {
    float m  = (float)zm[i];
    float lv = (float)zlv[i];
    z[i] = (_Float16)(m + __expf(0.5f * lv) * eps[i]);
  }
}

// ===========================================================================
// Per-pixel error: mean over 3 channels of (x - recon)^2
// ===========================================================================
__global__ void error_kernel(const float* __restrict__ x,
                             const _Float16* __restrict__ recon,
                             float* __restrict__ err, int npix) {
  int i = blockIdx.x * blockDim.x + threadIdx.x;
  if (i < npix) {
    float s = 0.0f;
#pragma unroll
    for (int c = 0; c < 3; ++c) {
      float d = x[(size_t)i * 3 + c] - (float)recon[(size_t)i * 3 + c];
      s += d * d;
    }
    err[i] = s * (1.0f / 3.0f);
  }
}

// ===========================================================================
// Otsu: zero bins, LDS-local histogram + global merge, scan/argmax, mask
// ===========================================================================
__global__ void hist_zero_kernel(float* __restrict__ hist) {
  int i = threadIdx.x;
  if (i < 2 * NBINS + 1) hist[i] = 0.0f;
}

__global__ void hist_kernel(const float* __restrict__ err,
                            float* __restrict__ counts,
                            float* __restrict__ sums, int n) {
  __shared__ float lc[NBINS];
  __shared__ float ls[NBINS];
  for (int i = threadIdx.x; i < NBINS; i += blockDim.x) { lc[i] = 0.f; ls[i] = 0.f; }
  __syncthreads();
  for (int i = blockIdx.x * blockDim.x + threadIdx.x; i < n;
       i += gridDim.x * blockDim.x) {
    float e = err[i];
    int b = (int)(e * 100.0f);
    b = b < 0 ? 0 : (b > NBINS - 1 ? NBINS - 1 : b);
    atomicAdd(&lc[b], 1.0f);
    atomicAdd(&ls[b], e);
  }
  __syncthreads();
  for (int i = threadIdx.x; i < NBINS; i += blockDim.x) {
    if (lc[i] != 0.0f) {
      atomicAdd(&counts[i], lc[i]);
      atomicAdd(&sums[i], ls[i]);
    }
  }
}

__global__ void otsu_kernel(const float* __restrict__ counts,
                            const float* __restrict__ sums,
                            float* __restrict__ thresh, float ntot) {
  if (threadIdx.x == 0 && blockIdx.x == 0) {
    float cge[NBINS], sge[NBINS];
    float c = 0.f, s = 0.f;
    for (int k = NBINS - 1; k >= 0; --k) {
      c += counts[k]; s += sums[k];
      cge[k] = c; sge[k] = s;
    }
    const float total_sum = sge[0];
    float best = -1.0f;
    int bi = 0;
    for (int k = 0; k < NBINS; ++k) {
      float cg  = cge[k];
      float cl  = ntot - cg;
      float mu1 = sge[k] / fmaxf(cg, 1.0f);
      float mu0 = (total_sum - sge[k]) / fmaxf(cl, 1.0f);
      float d   = mu1 - mu0;
      float sb  = (cg / ntot) * (cl / ntot) * d * d;
      if (sb > best) { best = sb; bi = k; }
    }
    *thresh = 0.01f * (float)bi;
  }
}

__global__ void mask_kernel(const float* __restrict__ err,
                            const float* __restrict__ thresh,
                            float* __restrict__ out, int n) {
  int i = blockIdx.x * blockDim.x + threadIdx.x;
  if (i < n) out[i] = err[i] > *thresh ? 1.0f : 0.0f;
}

// ===========================================================================
// Host-side orchestration
// ===========================================================================
extern "C" void kernel_launch(void* const* d_in, const int* in_sizes, int n_in,
                              void* d_out, int out_size, void* d_ws, size_t ws_size,
                              hipStream_t stream) {
  const int B = 4096;

  const float* x       = (const float*)d_in[0];
  const float* eps     = (const float*)d_in[1];
  const float* enc_w1  = (const float*)d_in[2];
  const float* enc_b1  = (const float*)d_in[3];
  const float* g1      = (const float*)d_in[4];
  const float* be1     = (const float*)d_in[5];
  const float* enc_w2  = (const float*)d_in[6];
  const float* enc_b2  = (const float*)d_in[7];
  const float* g2      = (const float*)d_in[8];
  const float* be2     = (const float*)d_in[9];
  const float* enc_w3  = (const float*)d_in[10];
  const float* enc_b3  = (const float*)d_in[11];
  const float* g3      = (const float*)d_in[12];
  const float* be3     = (const float*)d_in[13];
  const float* wzm     = (const float*)d_in[14];
  const float* bzm     = (const float*)d_in[15];
  const float* wzv     = (const float*)d_in[16];
  const float* bzv     = (const float*)d_in[17];
  const float* dec_wd  = (const float*)d_in[18];
  const float* dec_bd  = (const float*)d_in[19];
  const float* dec_w1  = (const float*)d_in[20];
  const float* dec_b1  = (const float*)d_in[21];
  const float* dg1     = (const float*)d_in[22];
  const float* db1     = (const float*)d_in[23];
  const float* dec_w2  = (const float*)d_in[24];
  const float* dec_b2  = (const float*)d_in[25];
  const float* dg2     = (const float*)d_in[26];
  const float* db2     = (const float*)d_in[27];
  const float* dec_w3  = (const float*)d_in[28];
  const float* dec_b3  = (const float*)d_in[29];
  const float* dg3     = (const float*)d_in[30];
  const float* db3     = (const float*)d_in[31];
  const float* dec_w4  = (const float*)d_in[32];
  const float* dec_b4  = (const float*)d_in[33];
  float* out = (float*)d_out;

  // ---- workspace carving (256B aligned) ----
  char* p = (char*)d_ws;
  auto alloc_h = [&](long nelem) -> _Float16* {
    _Float16* r = (_Float16*)p;
    p += ((nelem * 2 + 255) / 256) * 256;
    return r;
  };
  auto alloc_f = [&](long nelem) -> float* {
    float* r = (float*)p;
    p += ((nelem * 4 + 255) / 256) * 256;
    return r;
  };

  const long M1 = (long)B * 16 * 16;           // conv1 GEMM rows
  _Float16* a1   = alloc_h(M1 * 96);           // conv1 im2col, Kpad=96
  _Float16* bt1  = alloc_h(32L * 96);
  _Float16* bt2  = alloc_h(64L * 800);
  _Float16* bt3  = alloc_h(128L * 1600);
  _Float16* btzm = alloc_h(128L * 2048);
  _Float16* btzv = alloc_h(128L * 2048);
  _Float16* btwd = alloc_h(8192L * 128);
  _Float16* btd1 = alloc_h(128L * 128);
  _Float16* btd2 = alloc_h(64L * 1152);
  _Float16* btd3 = alloc_h(32L * 576);
  _Float16* btd4 = alloc_h(16L * 288);         // Npad 3->16
  _Float16* h1   = alloc_h((long)B * 16 * 16 * 32);
  _Float16* h2   = alloc_h((long)B * 8 * 8 * 64);
  _Float16* h3   = alloc_h((long)B * 4 * 4 * 128);
  _Float16* zm   = alloc_h((long)B * 128);
  _Float16* zlv  = alloc_h((long)B * 128);
  _Float16* z    = alloc_h((long)B * 128);
  _Float16* d0   = alloc_h((long)B * 8192);
  _Float16* d1   = alloc_h((long)B * 64 * 128);
  _Float16* d2   = alloc_h((long)B * 256 * 64);
  _Float16* d3   = alloc_h((long)B * 1024 * 32);
  _Float16* rec  = alloc_h((long)B * 1024 * 3);
  float*    err  = alloc_f((long)B * 1024);
  float*    hist = alloc_f(2 * NBINS + 1);
  float* counts = hist;
  float* sums   = hist + NBINS;
  float* thresh = hist + 2 * NBINS;

  auto wtrans = [&](const float* w, _Float16* bt, int K, int N, int Kpad, int Npad) {
    int tot = Npad * Kpad;
    wtrans_kernel<<<(tot + 255) / 256, 256, 0, stream>>>(w, bt, K, N, Kpad, Npad);
  };

  // ---- weight transposes (zero-padded [Npad, Kpad]) ----
  wtrans(enc_w1, bt1, 75, 32, 96, 32);
  wtrans(enc_w2, bt2, 800, 64, 800, 64);
  wtrans(enc_w3, bt3, 1600, 128, 1600, 128);
  wtrans(wzm, btzm, 2048, 128, 2048, 128);
  wtrans(wzv, btzv, 2048, 128, 2048, 128);
  wtrans(dec_wd, btwd, 128, 8192, 128, 8192);
  wtrans(dec_w1, btd1, 128, 128, 128, 128);
  wtrans(dec_w2, btd2, 1152, 64, 1152, 64);
  wtrans(dec_w3, btd3, 576, 32, 576, 32);
  wtrans(dec_w4, btd4, 288, 3, 288, 16);

  // ---- conv1: im2col + GEMM (M=1048576, K=96, N=32; MT=2, NT=2) ----
  {
    long tot = M1 * 96;
    im2col_conv1_kernel<<<(int)((tot + 255) / 256), 256, 0, stream>>>(x, a1, (int)tot);
    dim3 grid((int)((M1 / 16 + 15) / 16), 1);
    wmma_gemm_tpl<2, 2, 1, 1><<<grid, 256, 0, stream>>>(
        a1, bt1, enc_b1, g1, be1, h1, (int)M1, 96, 32);
  }

  // ---- conv2: 16x16x32 -> 8x8x64 (k5 s2; MT=1, NT=4) ----
  {
    int M = B * 64;
    dim3 grid((M / 16 + 7) / 8, 1);
    wmma_conv_tpl<16, 16, 32, 5, 5, 8, 8, 2, 2, 0, 1, 4, 1, 1>
        <<<grid, 256, 0, stream>>>(h1, bt2, enc_b2, g2, be2, h2, B, 64);
  }
  // ---- conv3: 8x8x64 -> 4x4x128 (k5 s2; MT=1, NT=4) ----
  {
    int M = B * 16;
    dim3 grid((M / 16 + 7) / 8, 2);
    wmma_conv_tpl<8, 8, 64, 5, 5, 4, 4, 2, 2, 0, 1, 4, 1, 1>
        <<<grid, 256, 0, stream>>>(h2, bt3, enc_b3, g3, be3, h3, B, 128);
  }

  // ---- latent heads: dense [4096,2048] x [2048,128] ----
  {
    dim3 grid((B / 16 + 7) / 8, 2);
    wmma_gemm_tpl<1, 4, 1, 0><<<grid, 256, 0, stream>>>(
        h3, btzm, bzm, nullptr, nullptr, zm, B, 2048, 128);
    wmma_gemm_tpl<1, 4, 1, 0><<<grid, 256, 0, stream>>>(
        h3, btzv, bzv, nullptr, nullptr, zlv, B, 2048, 128);
  }
  {
    int n = B * 128;
    sampling_kernel<<<(n + 255) / 256, 256, 0, stream>>>(zm, zlv, eps, z, n);
  }

  // ---- decoder dense: [4096,128] x [128,8192] ----
  {
    dim3 grid((B / 16 + 7) / 8, 128);
    wmma_gemm_tpl<1, 4, 1, 0><<<grid, 256, 0, stream>>>(
        z, btwd, dec_bd, nullptr, nullptr, d0, B, 128, 8192);
  }
  // ---- deconv1: 1x1, dense GEMM [B*64,128] x [128,128] ----
  {
    int M = B * 64;
    dim3 grid((M / 16 + 7) / 8, 2);
    wmma_gemm_tpl<1, 4, 1, 1><<<grid, 256, 0, stream>>>(
        d0, btd1, dec_b1, dg1, db1, d1, M, 128, 128);
  }
  // ---- deconv2: 8x8x128 -> 16x16x64 (k3 s2 convT; MT=1, NT=4) ----
  {
    int M = B * 256;
    dim3 grid((M / 16 + 7) / 8, 1);
    wmma_conv_tpl<8, 8, 128, 3, 3, 16, 16, 2, 1, 1, 1, 4, 1, 1>
        <<<grid, 256, 0, stream>>>(d1, btd2, dec_b2, dg2, db2, d2, B, 64);
  }
  // ---- deconv3: 16x16x64 -> 32x32x32 (k3 s2 convT; MT=2, NT=2) ----
  {
    int M = B * 1024;
    dim3 grid((M / 16 + 15) / 16, 1);
    wmma_conv_tpl<16, 16, 64, 3, 3, 32, 32, 2, 1, 1, 2, 2, 1, 1>
        <<<grid, 256, 0, stream>>>(d2, btd3, dec_b3, dg3, db3, d3, B, 32);
  }
  // ---- deconv4: 32x32x32 -> 32x32x3 (k3 s1 convT, sigmoid; MT=4, NT=1) ----
  {
    int M = B * 1024;
    dim3 grid((M / 16 + 31) / 32, 1);
    wmma_conv_tpl<32, 32, 32, 3, 3, 32, 32, 1, 1, 1, 4, 1, 2, 0>
        <<<grid, 256, 0, stream>>>(d3, btd4, dec_b4, nullptr, nullptr, rec, B, 3);
  }

  // ---- error + Otsu threshold + mask ----
  const int npix = B * 1024;
  error_kernel<<<(npix + 255) / 256, 256, 0, stream>>>(x, rec, err, npix);
  hist_zero_kernel<<<1, 256, 0, stream>>>(hist);
  hist_kernel<<<512, 256, 0, stream>>>(err, counts, sums, npix);
  otsu_kernel<<<1, 32, 0, stream>>>(counts, sums, thresh, (float)npix);
  mask_kernel<<<(npix + 255) / 256, 256, 0, stream>>>(err, thresh, out, npix);
  (void)in_sizes; (void)n_in; (void)out_size; (void)ws_size;
}
